// DeformableRefinementBlock_18777597018132
// MI455X (gfx1250) — compile-verified, matrix operand order checked by reading, not run on hardware
//
#include <hip/hip_runtime.h>

// ---------------------------------------------------------------------------
// Problem constants (reference: B=2, C=128, H=W=128)
// ---------------------------------------------------------------------------
#define BB   2
#define CC   128
#define HH   128
#define WW   128
#define HW   (HH * WW)          // 16384 = 1<<14
#define CIN0 160                // 130 (feat+flow) padded to 5 chunks of 32

typedef __attribute__((ext_vector_type(16))) _Float16 v16h;
typedef __attribute__((ext_vector_type(8)))  _Float16 v8h;
typedef __attribute__((ext_vector_type(8)))  float    v8f;

union H16 { v16h v; uint4 u[2]; };
union H8  { v8h  v; uint4 u;    };

static __device__ __forceinline__ v8f wmma_f16(const H16& a, const H16& b, v8f c) {
    return __builtin_amdgcn_wmma_f32_16x16x32_f16(false, a.v, false, b.v,
                                                  (short)0, c, false, false);
}

// ---------------------------------------------------------------------------
// Pack feat(NCHW f32) + flow(NCHW f32) -> X0 (B,H,W,160) f16, zero padded.
// ---------------------------------------------------------------------------
__global__ void pack_input_kernel(const float* __restrict__ feat,
                                  const float* __restrict__ flow,
                                  _Float16* __restrict__ X0) {
    int g = blockIdx.x * blockDim.x + threadIdx.x;        // one per 8-channel group
    const int total = BB * HW * (CIN0 / 8);
    if (g >= total) return;
    int c8  = (g % (CIN0 / 8)) * 8;
    int pix = g / (CIN0 / 8);
    int x = pix & (WW - 1);
    int y = (pix >> 7) & (HH - 1);
    int b = pix >> 14;
    H8 v;
    for (int j = 0; j < 8; ++j) {
        int c = c8 + j;
        float val = 0.f;
        if (c < CC)            val = feat[(((size_t)(b * CC + c)) << 14) + (y << 7) + x];
        else if (c < CC + 2)   val = flow[(((size_t)(b * 2 + (c - CC))) << 14) + (y << 7) + x];
        v.v[j] = (_Float16)val;
    }
    *(uint4*)(X0 + (size_t)pix * CIN0 + c8) = v.u;
}

// ---------------------------------------------------------------------------
// Repack f32 OIHW 3x3 weights into per-lane WMMA A-fragments (f16).
// Layout: [mtile][tap][chunk][lane(32)][e(16)]  -> lane loads 32B contiguous.
// A-frag mapping: lane l: m = l&15, base_k = (l>>4)*8; elem e: k = base_k + (e&7) + (e>>3)*16
// ---------------------------------------------------------------------------
__global__ void repack_w_kernel(const float* __restrict__ w, _Float16* __restrict__ wp,
                                int Cout, int Cin, int Mtiles, int Cchunks) {
    int g = blockIdx.x * blockDim.x + threadIdx.x;
    int total = Mtiles * 9 * Cchunks * 32;
    if (g >= total) return;
    int lane  = g & 31;
    int rest  = g >> 5;
    int chunk = rest % Cchunks; rest /= Cchunks;
    int tap   = rest % 9;
    int mtile = rest / 9;
    int o     = mtile * 16 + (lane & 15);
    int basek = (lane >> 4) * 8;
    H16 frag;
    for (int e = 0; e < 16; ++e) {
        int c = chunk * 32 + basek + (e & 7) + ((e >> 3) << 4);
        float val = (o < Cout && c < Cin) ? w[((size_t)o * Cin + c) * 9 + tap] : 0.f;
        frag.v[e] = (_Float16)val;
    }
    *(uint4*)(wp + (size_t)g * 16)     = frag.u[0];
    *(uint4*)(wp + (size_t)g * 16 + 8) = frag.u[1];
}

// ---------------------------------------------------------------------------
// 3x3 SAME conv via implicit GEMM + WMMA.  X: (B,H,W,CinPad) f16.
// One workgroup = 16-pixel row segment; wave w owns Cout tile [16w,16w+16).
// Output: f16 NHWC with stride CoutStride.  Optional ReLU, bias fused.
// ---------------------------------------------------------------------------
__global__ void conv3x3_wmma_kernel(const _Float16* __restrict__ X, int CinPad,
                                    const _Float16* __restrict__ Wp,
                                    const float* __restrict__ bias,
                                    _Float16* __restrict__ Y, int CoutStride,
                                    int doRelu) {
    extern __shared__ __align__(16) char smem_raw[];
    _Float16* lds = (_Float16*)smem_raw;

    const int x0 = blockIdx.x * 16, y = blockIdx.y, b = blockIdx.z;
    const int tid = threadIdx.x;
    const int CinVec = CinPad >> 3;

    // Stage 3x18xCinPad halo tile (zero SAME padding) with b128 copies.
    const int nvec = 3 * 18 * CinVec;
    for (int i = tid; i < nvec; i += blockDim.x) {
        int cv  = i % CinVec;
        int rc  = i / CinVec;
        int col = rc % 18;
        int r   = rc / 18;
        int gy = y + r - 1, gx = x0 + col - 1;
        uint4 v = {0u, 0u, 0u, 0u};
        if ((unsigned)gy < (unsigned)HH && (unsigned)gx < (unsigned)WW)
            v = *(const uint4*)(X + ((((size_t)(b << 7) + gy) << 7) + gx) * CinPad + (cv << 3));
        *(uint4*)(lds + (size_t)i * 8) = v;
    }
    __syncthreads();

    const int lane = tid & 31, mtile = tid >> 5;
    const int px = lane & 15, hi = lane >> 4;
    const int Cchunks = CinPad >> 5;
    v8f acc = {};

    for (int tap = 0; tap < 9; ++tap) {
        int r   = tap / 3;
        int col = px + (tap % 3);
        const _Float16* bbase = lds + (r * 18 + col) * CinPad + hi * 8;
        const _Float16* abase = Wp + ((size_t)(mtile * 9 + tap) * Cchunks) * 512 + lane * 16;
        for (int ch = 0; ch < Cchunks; ++ch) {
            H16 a, bf;
            a.u[0]  = *(const uint4*)(abase + ch * 512);
            a.u[1]  = *(const uint4*)(abase + ch * 512 + 8);
            bf.u[0] = *(const uint4*)(bbase + ch * 32);
            bf.u[1] = *(const uint4*)(bbase + ch * 32 + 16);
            acc = wmma_f16(a, bf, acc);
        }
    }

    // Epilogue: bias + relu, 8 consecutive channels per lane -> one b128 store.
    int obase = mtile * 16 + hi * 8;
    H8 outv;
    for (int r = 0; r < 8; ++r) {
        float v = acc[r] + bias[obase + r];
        if (doRelu) v = v > 0.f ? v : 0.f;
        outv.v[r] = (_Float16)v;
    }
    *(uint4*)(Y + ((((size_t)(b << 7) + y) << 7) + x0 + px) * CoutStride + obase) = outv.u;
}

// ---------------------------------------------------------------------------
// Head conv: same GEMM but writes f32 NCHW with CoutReal channels (Cout padded
// to Mtiles*16 with zero weights). Optionally fuses elementwise add (flow).
// ---------------------------------------------------------------------------
__global__ void conv3x3_head_kernel(const _Float16* __restrict__ X, int CinPad,
                                    const _Float16* __restrict__ Wp,
                                    const float* __restrict__ bias,
                                    float* __restrict__ Yn, int CoutReal,
                                    const float* __restrict__ addB) {
    extern __shared__ __align__(16) char smem_raw[];
    _Float16* lds = (_Float16*)smem_raw;

    const int x0 = blockIdx.x * 16, y = blockIdx.y, b = blockIdx.z;
    const int tid = threadIdx.x;
    const int CinVec = CinPad >> 3;

    const int nvec = 3 * 18 * CinVec;
    for (int i = tid; i < nvec; i += blockDim.x) {
        int cv  = i % CinVec;
        int rc  = i / CinVec;
        int col = rc % 18;
        int r   = rc / 18;
        int gy = y + r - 1, gx = x0 + col - 1;
        uint4 v = {0u, 0u, 0u, 0u};
        if ((unsigned)gy < (unsigned)HH && (unsigned)gx < (unsigned)WW)
            v = *(const uint4*)(X + ((((size_t)(b << 7) + gy) << 7) + gx) * CinPad + (cv << 3));
        *(uint4*)(lds + (size_t)i * 8) = v;
    }
    __syncthreads();

    const int lane = tid & 31, mtile = tid >> 5;
    const int px = lane & 15, hi = lane >> 4;
    const int Cchunks = CinPad >> 5;
    v8f acc = {};

    for (int tap = 0; tap < 9; ++tap) {
        int r   = tap / 3;
        int col = px + (tap % 3);
        const _Float16* bbase = lds + (r * 18 + col) * CinPad + hi * 8;
        const _Float16* abase = Wp + ((size_t)(mtile * 9 + tap) * Cchunks) * 512 + lane * 16;
        for (int ch = 0; ch < Cchunks; ++ch) {
            H16 a, bf;
            a.u[0]  = *(const uint4*)(abase + ch * 512);
            a.u[1]  = *(const uint4*)(abase + ch * 512 + 8);
            bf.u[0] = *(const uint4*)(bbase + ch * 32);
            bf.u[1] = *(const uint4*)(bbase + ch * 32 + 16);
            acc = wmma_f16(a, bf, acc);
        }
    }

    for (int r = 0; r < 8; ++r) {
        int o = mtile * 16 + hi * 8 + r;
        if (o < CoutReal) {
            float v = acc[r] + bias[o];
            size_t idx = (((size_t)(b * CoutReal + o)) << 14) + (y << 7) + x0 + px;
            if (addB) v += addB[idx];
            Yn[idx] = v;
        }
    }
}

// ---------------------------------------------------------------------------
// Deformable 3x3 conv, 128->128.  Per tap: 256 threads bilinear-sample a
// 128ch x 16px f16 tile into LDS, then 8 waves run 4 WMMAs each (K=128),
// accumulating across all 9 taps in registers.
// X0: (B,H,W,160) f16 (feat = ch 0..127).  OFF: (B,18,H,W) f32.
// ---------------------------------------------------------------------------
__global__ void deform_conv_wmma_kernel(const _Float16* __restrict__ X0,
                                        const float* __restrict__ OFF,
                                        const _Float16* __restrict__ Wp,
                                        const float* __restrict__ dbias,
                                        _Float16* __restrict__ Y) {
    __shared__ _Float16 samp[16 * 128];   // [pixel][channel]

    const int x0 = blockIdx.x * 16, y = blockIdx.y, b = blockIdx.z;
    const int tid = threadIdx.x;
    const int spx = tid >> 4;          // sampled pixel this thread fills
    const int c8  = (tid & 15) * 8;    // 8-channel group
    const int x   = x0 + spx;

    const int lane = tid & 31, mtile = tid >> 5;
    const int px = lane & 15, hi = lane >> 4;
    v8f acc = {};

    for (int tap = 0; tap < 9; ++tap) {
        __syncthreads();   // previous tap's WMMA reads of samp are done

        // ---- bilinear sample 8 channels at (y+dy+offy, x+dx+offx) ----
        float offy = OFF[(((size_t)(b * 18 + 2 * tap))     << 14) + (y << 7) + x];
        float offx = OFF[(((size_t)(b * 18 + 2 * tap + 1)) << 14) + (y << 7) + x];
        float fy = (float)y + (float)(tap / 3) - 1.f + offy;
        float fx = (float)x + (float)(tap % 3) - 1.f + offx;
        float y0f = floorf(fy), x0f = floorf(fx);
        float ly = fy - y0f, lx = fx - x0f;
        int iy = (int)y0f, ix = (int)x0f;
        float wgt[4] = {(1.f - ly) * (1.f - lx), (1.f - ly) * lx,
                        ly * (1.f - lx),          ly * lx};
        float av[8] = {0, 0, 0, 0, 0, 0, 0, 0};
        for (int corner = 0; corner < 4; ++corner) {
            int cy = iy + (corner >> 1);
            int cx = ix + (corner & 1);
            if ((unsigned)cy < (unsigned)HH && (unsigned)cx < (unsigned)WW) {
                H8 v;
                v.u = *(const uint4*)(X0 + ((((size_t)(b << 7) + cy) << 7) + cx) * CIN0 + c8);
                float w = wgt[corner];
                for (int j = 0; j < 8; ++j) av[j] += w * (float)v.v[j];
            }
        }
        H8 s;
        for (int j = 0; j < 8; ++j) s.v[j] = (_Float16)av[j];
        *(uint4*)(samp + spx * 128 + c8) = s.u;
        __syncthreads();

        // ---- 4 WMMAs over K=128 for this tap ----
        const _Float16* abase = Wp + ((size_t)(mtile * 9 + tap) * 4) * 512 + lane * 16;
        const _Float16* bbase = samp + px * 128 + hi * 8;
        for (int ch = 0; ch < 4; ++ch) {
            H16 a, bf;
            a.u[0]  = *(const uint4*)(abase + ch * 512);
            a.u[1]  = *(const uint4*)(abase + ch * 512 + 8);
            bf.u[0] = *(const uint4*)(bbase + ch * 32);
            bf.u[1] = *(const uint4*)(bbase + ch * 32 + 16);
            acc = wmma_f16(a, bf, acc);
        }
    }

    int obase = mtile * 16 + hi * 8;
    H8 outv;
    for (int r = 0; r < 8; ++r)
        outv.v[r] = (_Float16)(acc[r] + dbias[obase + r]);
    *(uint4*)(Y + ((((size_t)(b << 7) + y) << 7) + x0 + px) * 128 + obase) = outv.u;
}

// ---------------------------------------------------------------------------
// Host-side orchestration
// ---------------------------------------------------------------------------
extern "C" void kernel_launch(void* const* d_in, const int* in_sizes, int n_in,
                              void* d_out, int out_size, void* d_ws, size_t ws_size,
                              hipStream_t stream) {
    (void)in_sizes; (void)n_in; (void)out_size; (void)ws_size;

    const float* feat    = (const float*)d_in[0];
    const float* flow    = (const float*)d_in[1];
    const float* off_w1  = (const float*)d_in[2];
    const float* off_b1  = (const float*)d_in[3];
    const float* off_w2  = (const float*)d_in[4];
    const float* off_b2  = (const float*)d_in[5];
    const float* dweight = (const float*)d_in[6];
    const float* dbias   = (const float*)d_in[7];
    const float* fh_w1   = (const float*)d_in[8];
    const float* fh_b1   = (const float*)d_in[9];
    const float* fh_w2   = (const float*)d_in[10];
    const float* fh_b2   = (const float*)d_in[11];
    float* out = (float*)d_out;

    char* ws = (char*)d_ws;
    size_t off = 0;
    auto alloc = [&](size_t bytes) -> char* {
        char* p = ws + off;
        off = (off + bytes + 255) & ~(size_t)255;
        return p;
    };

    _Float16* X0  = (_Float16*)alloc((size_t)BB * HW * CIN0 * 2);   // 10 MB
    _Float16* T1  = (_Float16*)alloc((size_t)BB * HW * 64 * 2);     // conv1 out (relu)
    float*    OFS = (float*)   alloc((size_t)BB * 18 * HW * 4);     // offset (NCHW)
    _Float16* DO_ = (_Float16*)alloc((size_t)BB * HW * 128 * 2);    // deform out
    _Float16* H1  = (_Float16*)alloc((size_t)BB * HW * 64 * 2);     // fh1 out (relu)
    _Float16* W1P = (_Float16*)alloc((size_t)4 * 9 * 5 * 512 * 2);  // conv1 weights
    _Float16* W2P = (_Float16*)alloc((size_t)2 * 9 * 2 * 512 * 2);  // conv2 weights
    _Float16* WDP = (_Float16*)alloc((size_t)8 * 9 * 4 * 512 * 2);  // deform weights
    _Float16* WF1 = (_Float16*)alloc((size_t)4 * 9 * 4 * 512 * 2);  // fh1 weights
    _Float16* WF2 = (_Float16*)alloc((size_t)1 * 9 * 2 * 512 * 2);  // fh2 weights

    // 1) pack feat+flow into padded f16 NHWC
    {
        int total = BB * HW * (CIN0 / 8);
        pack_input_kernel<<<(total + 255) / 256, 256, 0, stream>>>(feat, flow, X0);
    }

    // 2) repack all weights into WMMA A-fragment layout
    auto repack = [&](const float* w, _Float16* wp, int Cout, int Cin, int Mtiles, int Cchunks) {
        int total = Mtiles * 9 * Cchunks * 32;
        repack_w_kernel<<<(total + 255) / 256, 256, 0, stream>>>(w, wp, Cout, Cin, Mtiles, Cchunks);
    };
    repack(off_w1,  W1P,  64, 130, 4, 5);
    repack(off_w2,  W2P,  18,  64, 2, 2);
    repack(dweight, WDP, 128, 128, 8, 4);
    repack(fh_w1,   WF1,  64, 128, 4, 4);
    repack(fh_w2,   WF2,   2,  64, 1, 2);

    dim3 grid(WW / 16, HH, BB);

    // 3) conv1: (B,H,W,160) -> relu -> (B,H,W,64) f16
    conv3x3_wmma_kernel<<<grid, 4 * 32, 3 * 18 * CIN0 * 2, stream>>>(
        X0, CIN0, W1P, off_b1, T1, 64, 1);

    // 4) conv2 head: (B,H,W,64) -> offset (B,18,H,W) f32
    conv3x3_head_kernel<<<grid, 2 * 32, 3 * 18 * 64 * 2, stream>>>(
        T1, 64, W2P, off_b2, OFS, 18, nullptr);

    // 5) deformable conv: feat (in X0) + offset -> (B,H,W,128) f16
    deform_conv_wmma_kernel<<<grid, 256, 0, stream>>>(X0, OFS, WDP, dbias, DO_);

    // 6) fh1: (B,H,W,128) -> relu -> (B,H,W,64) f16
    conv3x3_wmma_kernel<<<grid, 4 * 32, 3 * 18 * 128 * 2, stream>>>(
        DO_, 128, WF1, fh_b1, H1, 64, 1);

    // 7) fh2 head: (B,H,W,64) -> delta_flow; fuse "+ flow" -> d_out (B,2,H,W) f32
    conv3x3_head_kernel<<<grid, 1 * 32, 3 * 18 * 64 * 2, stream>>>(
        H1, 64, WF2, fh_b2, out, 2, flow);
}
